// MNIST_GNN_40527311405184
// MI455X (gfx1250) — matrix operand order repaired and checked by reading, not compiled, not run on hardware
//
#include <hip/hip_runtime.h>
#include <hip/hip_bf16.h>

typedef __bf16 bf16;
typedef __attribute__((ext_vector_type(16))) __bf16 v16bf;
typedef __attribute__((ext_vector_type(8)))  __bf16 v8bf;
typedef __attribute__((ext_vector_type(8)))  float  v8f;
typedef __attribute__((ext_vector_type(4)))  unsigned uv4i;
typedef __attribute__((ext_vector_type(4)))  int      iv4;
typedef __attribute__((ext_vector_type(8)))  int      iv8;

#define HDIM 64
#define GIN_L 4
#define BNEPS 1e-5f
#define ROWS_PER_BLOCK 128

__device__ __forceinline__ unsigned short f2bf_bits(float f) {
  unsigned u = __builtin_bit_cast(unsigned, f);
  unsigned r = u + 0x7FFFu + ((u >> 16) & 1u);   // round-to-nearest-even
  return (unsigned short)(r >> 16);
}
__device__ __forceinline__ bf16 f2bf(float f) {
  return __builtin_bit_cast(bf16, f2bf_bits(f));
}

// Issue one TDM load of a [rows x 64] f32 tile (row stride 64) into LDS.
// D# per CDNA5 ISA ch.8: group0 = {count/type/addrs}, group1 = {dims/tile}.
// OOB rows (y >= tensor_dim1) read as zero -> free tail-block zero fill.
__device__ __forceinline__ void tdm_load_tile_f32(unsigned lds_off,
                                                  const float* gptr,
                                                  unsigned rows_remaining) {
  unsigned long long ga = (unsigned long long)(uintptr_t)gptr;
  uv4i g0;
  g0[0] = 1u;                                  // count=1, is_restore=0, gather off
  g0[1] = lds_off;                             // lds_addr (bytes)
  g0[2] = (unsigned)(ga & 0xFFFFFFFFull);      // global_addr[31:0]
  g0[3] = (unsigned)(ga >> 32) | 0x80000000u;  // global_addr[56:32] | type=2 (bits 127:126 = 10b)
  iv8 g1;
  g1[0] = (int)(2u << 16);                     // workgroup_mask=0, data_size=2 (4 bytes)
  g1[1] = (int)(64u << 16);                    // atomic_barrier=0 | tensor_dim0[15:0]=64
  g1[2] = (int)((rows_remaining & 0xFFFFu) << 16);        // tensor_dim0[31:16]=0 | tensor_dim1[15:0]
  g1[3] = (int)((rows_remaining >> 16) | (64u << 16));    // tensor_dim1[31:16] | tile_dim0=64
  g1[4] = (int)ROWS_PER_BLOCK;                 // tile_dim1=128, tile_dim2=0
  g1[5] = 64;                                  // tensor_dim0_stride[31:0] = 64 elements
  g1[6] = 0;                                   // stride0[47:32]=0, dim1_stride[15:0]=0
  g1[7] = 0;
  iv4 gz = {0, 0, 0, 0};
#if defined(__clang_major__) && (__clang_major__ >= 23)
  iv8 gz8 = {0, 0, 0, 0, 0, 0, 0, 0};
  __builtin_amdgcn_tensor_load_to_lds(g0, g1, gz, gz, gz8, 0);
#else
  __builtin_amdgcn_tensor_load_to_lds(g0, g1, gz, gz, 0);
#endif
}

__global__ void zero_kernel(float* __restrict__ p, int n) {
  int i = blockIdx.x * blockDim.x + threadIdx.x;
  if (i < n) p[i] = 0.0f;
}

__global__ void count_kernel(const int* __restrict__ batch, float* __restrict__ counts, int n) {
  int i = blockIdx.x * blockDim.x + threadIdx.x;
  if (i < n) atomicAdd(&counts[batch[i]], 1.0f);
}

// Fused GEMM: Y[n,64] = prologue( A[n,64] ) @ W[64,64] + bias
// mode 0: a = A0                       (embedding: x @ W_emb)
// mode 1: a = A0 + A1                  (GIN aggregate: (h+agg) @ W1)
// mode 2: a = relu(A0*scl[c]+shf[c])   (BN+ReLU fused into A-load: z @ W2)
__global__ __launch_bounds__(256) void gemm64_kernel(
    int mode, const float* __restrict__ A0, const float* __restrict__ A1,
    const float* __restrict__ scl, const float* __restrict__ shf,
    const float* __restrict__ W, const float* __restrict__ bias,
    float* __restrict__ Y, int nrows) {
  __shared__ __align__(16) float    ldsAf[ROWS_PER_BLOCK * HDIM];      // 32 KB TDM dest (A0 tile)
  __shared__ __align__(16) float    ldsAg[ROWS_PER_BLOCK * HDIM];      // 32 KB TDM dest (A1 tile)
  __shared__ __align__(16) unsigned ldsA32[ROWS_PER_BLOCK * HDIM / 2]; // packed bf16, 16 KB
  __shared__ __align__(16) bf16     ldsWt[HDIM * HDIM];                // W^T: [n][k], 8 KB
  __shared__ float ldsB[HDIM];
  __shared__ float ldsScl[HDIM];
  __shared__ float ldsShf[HDIM];

  const bf16* ldsA = (const bf16*)ldsA32;
  const int tid = threadIdx.x;
  const int rowbase = blockIdx.x * ROWS_PER_BLOCK;

  // Wave 0 kicks off the async tensor DMA(s) for the A tile(s), then waits
  // on TENSORcnt. Wave-uniform branch: TDM ignores EXEC, so it must be
  // branched around by the other waves, not masked.
  if (tid < 32) {
    const unsigned rows_rem = (unsigned)(nrows - rowbase);
    tdm_load_tile_f32((unsigned)(uintptr_t)&ldsAf[0],
                      A0 + (size_t)rowbase * HDIM, rows_rem);
    if (mode == 1)
      tdm_load_tile_f32((unsigned)(uintptr_t)&ldsAg[0],
                        A1 + (size_t)rowbase * HDIM, rows_rem);
    __builtin_amdgcn_s_wait_tensorcnt(0);
  }

  // Meanwhile: stage transposed weights (f32 -> bf16), bias, BN params.
  #pragma unroll
  for (int p = tid; p < HDIM * HDIM; p += 256) {
    int k = p >> 6, n = p & 63;
    ldsWt[n * HDIM + k] = f2bf(W[p]);
  }
  if (tid < HDIM) {
    ldsB[tid] = bias[tid];
    if (mode == 2) { ldsScl[tid] = scl[tid]; ldsShf[tid] = shf[tid]; }
  }
  __syncthreads();   // TDM data + weights visible to all waves

  // LDS->LDS fixup: fused elementwise prologue + f32 -> packed bf16x2.
  #pragma unroll
  for (int p = tid; p < ROWS_PER_BLOCK * HDIM / 2; p += 256) {
    int cp = (p & 31) * 2;      // column pair (row-major: elem 2p -> col (2p)&63)
    float2 v = *(const float2*)(ldsAf + 2 * p);
    float v0 = v.x, v1 = v.y;
    if (mode == 1) {
      float2 w = *(const float2*)(ldsAg + 2 * p);
      v0 += w.x; v1 += w.y;
    } else if (mode == 2) {
      v0 = fmaxf(v0 * ldsScl[cp]     + ldsShf[cp],     0.0f);
      v1 = fmaxf(v1 * ldsScl[cp + 1] + ldsShf[cp + 1], 0.0f);
    }
    unsigned lo = (unsigned)f2bf_bits(v0);
    unsigned hi = (unsigned)f2bf_bits(v1);
    ldsA32[p] = lo | (hi << 16);
  }
  __syncthreads();

  const int lane = tid & 31;
  const int wave = tid >> 5;
  const int wrow = wave * 16;      // this wave's 16-row tile within the slab
  const int m    = lane & 15;
  const int hi   = lane >> 4;      // lane half (wave32)

  v8f acc[4] = {};                 // 4 N-tiles of 16x16 f32

  #pragma unroll
  for (int kc = 0; kc < 2; ++kc) { // K = 64 = 2 chunks of 32
    // A fragment (16x32 bf16): lanes 0-15 -> K {0..15}, lanes 16-31 -> K {16..31}
    const bf16* arow = &ldsA[(wrow + m) * HDIM + kc * 32 + hi * 16];
    v8bf a_lo = *(const v8bf*)(arow);
    v8bf a_hi = *(const v8bf*)(arow + 8);
    v16bf a = __builtin_shufflevector(a_lo, a_hi,
        0, 1, 2, 3, 4, 5, 6, 7, 8, 9, 10, 11, 12, 13, 14, 15);

    #pragma unroll
    for (int t = 0; t < 4; ++t) {
      // B fragment (32x16 bf16): lane n = t*16+m holds K kbB..kbB+15 contiguous in W^T
      const int n   = t * 16 + m;
      const int kbB = kc * 32 + hi * 16;
      v16bf b = *(const v16bf*)(ldsWt + n * HDIM + kbB);  // 2x ds_load_b128

      acc[t] = __builtin_amdgcn_wmma_f32_16x16x32_bf16(
          false, a, false, b, (short)0, acc[t], false, false);
    }
  }

  // Store: C/D layout — VGPR r holds M=r (lanes 0-15) / M=r+8 (lanes 16-31), N=lane&15
  if (rowbase + ROWS_PER_BLOCK <= nrows) {
    // Fast path: whole slab in range, no per-element predication.
    #pragma unroll
    for (int t = 0; t < 4; ++t) {
      const int   col = t * 16 + m;
      const float bv  = ldsB[col];
      #pragma unroll
      for (int r = 0; r < 8; ++r) {
        int grow = rowbase + wrow + r + hi * 8;
        Y[(size_t)grow * HDIM + col] = acc[t][r] + bv;
      }
    }
  } else {
    #pragma unroll
    for (int t = 0; t < 4; ++t) {
      const int   col = t * 16 + m;
      const float bv  = ldsB[col];
      #pragma unroll
      for (int r = 0; r < 8; ++r) {
        int grow = rowbase + wrow + r + hi * 8;
        if (grow < nrows) Y[(size_t)grow * HDIM + col] = acc[t][r] + bv;
      }
    }
  }
}

// agg[dst] += h[src] over edges; one wave per edge, 2 feats per lane
__global__ void scatter_kernel(const float* __restrict__ h, const int* __restrict__ src,
                               const int* __restrict__ dst, float* __restrict__ agg, int E) {
  long long t = (long long)blockIdx.x * blockDim.x + threadIdx.x;
  int e = (int)(t >> 5), lane = (int)(t & 31);
  if (e < E) {
    int s = src[e], d = dst[e];
    const float2 v = *(const float2*)(h + (size_t)s * HDIM + lane * 2);
    float* ap = agg + (size_t)d * HDIM + lane * 2;
    atomicAdd(ap, v.x);
    atomicAdd(ap + 1, v.y);
  }
}

// per-feature sum & sumsq over rows -> stats[0..63]=sum, stats[64..127]=sumsq
__global__ __launch_bounds__(256) void stats_kernel(const float* __restrict__ y,
                                                    float* __restrict__ stats, int nrows) {
  __shared__ float s1[256], s2[256];
  int c  = threadIdx.x & 63;
  int rg = threadIdx.x >> 6;                 // 4 row-groups per block
  float a = 0.0f, b = 0.0f;
  for (int row = blockIdx.x * 4 + rg; row < nrows; row += gridDim.x * 4) {
    float v = y[(size_t)row * HDIM + c];
    a += v; b += v * v;
  }
  s1[threadIdx.x] = a; s2[threadIdx.x] = b;
  __syncthreads();
  if (threadIdx.x < 64) {
    a = s1[threadIdx.x] + s1[threadIdx.x + 64] + s1[threadIdx.x + 128] + s1[threadIdx.x + 192];
    b = s2[threadIdx.x] + s2[threadIdx.x + 64] + s2[threadIdx.x + 128] + s2[threadIdx.x + 192];
    atomicAdd(&stats[c], a);
    atomicAdd(&stats[64 + c], b);
  }
}

__global__ void bn_finalize_kernel(const float* __restrict__ stats,
                                   const float* __restrict__ gamma, const float* __restrict__ beta,
                                   float* __restrict__ scl, float* __restrict__ shf, int nrows) {
  int c = threadIdx.x;
  if (c < HDIM) {
    float inv = 1.0f / (float)nrows;
    float mu  = stats[c] * inv;
    float var = stats[64 + c] * inv - mu * mu;
    float s   = gamma[c] * rsqrtf(var + BNEPS);
    scl[c] = s;
    shf[c] = beta[c] - mu * s;
  }
}

__global__ void pool_kernel(const float* __restrict__ h, const int* __restrict__ batch,
                            float* __restrict__ pooled, int rep, int n) {
  int t = blockIdx.x * blockDim.x + threadIdx.x;
  int node = t >> 6, c = t & 63;
  if (node < n) {
    int g = batch[node];
    atomicAdd(&pooled[(size_t)g * (HDIM * (GIN_L + 1)) + rep * HDIM + c],
              h[(size_t)node * HDIM + c]);
  }
}

__global__ void out_kernel(const float* __restrict__ pooled, const float* __restrict__ counts,
                           float* __restrict__ out, int total) {
  int i = blockIdx.x * blockDim.x + threadIdx.x;
  if (i < total) out[i] = pooled[i] / fmaxf(counts[i / (HDIM * (GIN_L + 1))], 1.0f);
}

extern "C" void kernel_launch(void* const* d_in, const int* in_sizes, int n_in,
                              void* d_out, int out_size, void* d_ws, size_t ws_size,
                              hipStream_t stream) {
  const float* x     = (const float*)d_in[0];
  const int*   ei    = (const int*)d_in[1];
  const int*   batch = (const int*)d_in[2];
  const float* W_emb = (const float*)d_in[3];
  const float* b_emb = (const float*)d_in[4];
  const float* W1    = (const float*)d_in[5];
  const float* b1    = (const float*)d_in[6];
  const float* gamma = (const float*)d_in[7];
  const float* beta  = (const float*)d_in[8];
  const float* W2    = (const float*)d_in[9];
  const float* b2    = (const float*)d_in[10];

  const int N = in_sizes[2];
  const int E = in_sizes[1] / 2;
  const int G = out_size / (HDIM * (GIN_L + 1));
  const int* src = ei;
  const int* dst = ei + E;

  float* ws = (float*)d_ws;
  const size_t nh = (size_t)N * HDIM;
  float* h      = ws;                       // [N,64]
  float* agg    = h + nh;                   // [N,64]
  float* y      = agg + nh;                 // [N,64]
  float* pooled = y + nh;                   // [G,320]
  float* counts = pooled + (size_t)G * (HDIM * (GIN_L + 1)); // [G]
  float* stats  = counts + G;               // [128]
  float* sclB   = stats + 128;              // [64]
  float* shfB   = sclB + 64;                // [64]

  dim3 b256(256);
  const int gemmGrid = (N + ROWS_PER_BLOCK - 1) / ROWS_PER_BLOCK;
  const int poolGrid = (int)((nh + 255) / 256);
  const int scatGrid = (int)(((size_t)E * 32 + 255) / 256);
  const int ptotal   = G * HDIM * (GIN_L + 1);

  zero_kernel<<<(ptotal + 255) / 256, b256, 0, stream>>>(pooled, ptotal);
  zero_kernel<<<(G + 255) / 256, b256, 0, stream>>>(counts, G);
  count_kernel<<<(N + 255) / 256, b256, 0, stream>>>(batch, counts, N);

  // embedding: h = x @ W_emb + b_emb
  gemm64_kernel<<<gemmGrid, b256, 0, stream>>>(0, x, nullptr, nullptr, nullptr,
                                               W_emb, b_emb, h, N);
  pool_kernel<<<poolGrid, b256, 0, stream>>>(h, batch, pooled, 0, N);

  for (int i = 0; i < GIN_L; ++i) {
    zero_kernel<<<(int)((nh + 255) / 256), b256, 0, stream>>>(agg, (int)nh);
    scatter_kernel<<<scatGrid, b256, 0, stream>>>(h, src, dst, agg, E);

    // y = (h + agg) @ W1[i] + b1[i]
    gemm64_kernel<<<gemmGrid, b256, 0, stream>>>(1, h, agg, nullptr, nullptr,
                                                 W1 + (size_t)i * HDIM * HDIM,
                                                 b1 + i * HDIM, y, N);
    // batch-norm stats over y
    zero_kernel<<<1, b256, 0, stream>>>(stats, 128);
    stats_kernel<<<512, b256, 0, stream>>>(y, stats, N);
    bn_finalize_kernel<<<1, 64, 0, stream>>>(stats, gamma + i * HDIM, beta + i * HDIM,
                                             sclB, shfB, N);
    // h = relu(bn(y)) @ W2[i] + b2[i]   (BN+ReLU fused into A staging)
    gemm64_kernel<<<gemmGrid, b256, 0, stream>>>(2, y, nullptr, sclB, shfB,
                                                 W2 + (size_t)i * HDIM * HDIM,
                                                 b2 + i * HDIM, h, N);
    pool_kernel<<<poolGrid, b256, 0, stream>>>(h, batch, pooled, i + 1, N);
  }

  out_kernel<<<(out_size + 255) / 256, b256, 0, stream>>>(pooled, counts,
                                                          (float*)d_out, out_size);
}
